// WindowAttention3D_18408229830980
// MI455X (gfx1250) — matrix operand order repaired and checked by reading, not compiled, not run on hardware
//
#include <hip/hip_runtime.h>

// ---- CDNA5 types ------------------------------------------------------------
typedef __attribute__((ext_vector_type(16))) __bf16       v16bf;
typedef __attribute__((ext_vector_type(8)))  float        v8f;
typedef __attribute__((ext_vector_type(4)))  unsigned int v4u;
typedef __attribute__((ext_vector_type(8)))  int          v8i;
typedef __attribute__((ext_vector_type(4)))  int          v4i;

// LDS map (bytes):
//   [0,      32768)  xs bf16 [128][128]   phase1 A-source; reused as p [112][128]
//   [32768,  98304)  qk bf16 [128][256]   (phase0: aliased as x f32 staging [128][128])
//   [98304, 131072)  vT bf16 [128ch][128tok]
//   [131072,159744)  obuf bf16 [112][128]
#define LDS_BYTES 159744

// A fragment (16x32 bf16, MxK), source row-major [row][k], stride ld.
// lanes 0-15: M=lane, K=0..15; lanes 16-31: M=lane-16, K=16..31 (matches load_bT)
__device__ __forceinline__ v16bf load_a(const __bf16* p0, int row0, int ld, int k0) {
  const int lane = threadIdx.x & 31;
  const int m    = lane & 15;
  const int h8   = (lane >> 4) << 4;
  const __bf16* p = p0 + (row0 + m) * ld + k0 + h8;
  v16bf a;
#pragma unroll
  for (int j = 0; j < 8; ++j) { a[j] = p[j]; a[j + 8] = p[j + 8]; }
  return a;
}

// B fragment (32x16, KxN) taken from an N-major source M[n][k] (B = M^T):
// lane n=lane&15, K half = 16*(lane>>4); 16 contiguous bf16 per lane (2x b128).
__device__ __forceinline__ v16bf load_bT(const __bf16* p0, int row0, int ld, int k0) {
  const int lane = threadIdx.x & 31;
  const int n    = lane & 15;
  const int kh   = (lane >> 4) << 4;
  const __bf16* p = p0 + (row0 + n) * ld + k0 + kh;
  v16bf b;
#pragma unroll
  for (int e = 0; e < 16; ++e) b[e] = p[e];
  return b;
}

// Store 16x16 f32 D tile as bf16, row-major [row][col] stride ld.
// C/D layout: lanes 0-15: M=r; lanes 16-31: M=8+r; N=lane&15.
__device__ __forceinline__ void store_d_bf16(__bf16* p0, int row0, int col0, int ld,
                                             const v8f& d) {
  const int lane = threadIdx.x & 31;
  const int n    = lane & 15;
  const int mo   = (lane >> 4) << 3;
#pragma unroll
  for (int r = 0; r < 8; ++r)
    p0[(row0 + mo + r) * ld + col0 + n] = (__bf16)d[r];
}

// Store D tile TRANSPOSED into an N-major buffer t[n][row] stride ld
// (per-lane the 8 row-addresses are contiguous -> coalescible b16 stores).
__device__ __forceinline__ void store_d_bf16_T(__bf16* p0, int row0, int col0, int ld,
                                               const v8f& d) {
  const int lane = threadIdx.x & 31;
  const int n    = lane & 15;
  const int mo   = (lane >> 4) << 3;
  __bf16* p = p0 + (col0 + n) * ld + row0 + mo;
#pragma unroll
  for (int r = 0; r < 8; ++r) p[r] = (__bf16)d[r];
}

__device__ __forceinline__ v8f wmma_bf16(const v16bf& a, const v16bf& b, const v8f& c) {
  return __builtin_amdgcn_wmma_f32_16x16x32_bf16(false, a, false, b, (short)0, c,
                                                 false, false);
}

// ---- prep: weights fp32 -> bf16, TRANSPOSED to N-major [n][k] ---------------
__global__ void __launch_bounds__(256) wa3d_prep(const float* __restrict__ qkv_w,
                                                 const float* __restrict__ proj_w,
                                                 __bf16* __restrict__ wqT,
                                                 __bf16* __restrict__ wpT) {
  int i = blockIdx.x * blockDim.x + threadIdx.x;
  if (i < 128 * 384) {
    int k = i / 384, n = i % 384;              // qkv_w[k][n]
    wqT[n * 128 + k] = (__bf16)qkv_w[i];
  }
  if (i < 128 * 128) {
    int k = i >> 7, n = i & 127;               // proj_w[k][n]
    wpT[n * 128 + k] = (__bf16)proj_w[i];
  }
}

// ---- main: one workgroup (8 wave32) per window ------------------------------
__global__ void __launch_bounds__(256) wa3d_kernel(const float* __restrict__ x,
                                                   const float* __restrict__ qkv_b,
                                                   const float* __restrict__ proj_b,
                                                   const __bf16* __restrict__ wqT,
                                                   const __bf16* __restrict__ wpT,
                                                   float* __restrict__ out) {
  extern __shared__ char smem[];
  __bf16* xs   = (__bf16*)smem;               // [128][128]; reused as p [112][128]
  __bf16* qk   = (__bf16*)(smem + 32768);     // [128][256] q|k
  __bf16* vT   = (__bf16*)(smem + 98304);     // [128ch][128tok]
  __bf16* obuf = (__bf16*)(smem + 131072);    // [112][128]
  __bf16* pbuf = xs;

  const int b    = blockIdx.x;
  const int tid  = threadIdx.x;
  const int wave = tid >> 5;
  const int lane = tid & 31;
  const int nlo  = lane & 15;
  const int mo   = (lane >> 4) << 3;

  // ---- Phase 0: TDM DMA of x window (98x128 f32, zero-padded to 128 rows by
  //      tensor OOB semantics) into LDS staging, then convert to bf16. ----
  if (wave == 0) {
    unsigned long long ga = (unsigned long long)(const void*)(x + (size_t)b * 98 * 128);
    unsigned ldsa = (unsigned)(unsigned long long)(smem + 32768);
    v4u g0;
    g0[0] = 1u;                                            // count=1 (user D#)
    g0[1] = ldsa;                                          // lds_addr
    g0[2] = (unsigned)ga;                                  // global_addr[31:0]
    g0[3] = (unsigned)((ga >> 32) & 0x1FFFFFFu) | (2u << 30); // addr[56:32]|type=2
    v8i g1;
    g1[0] = 0x00020000;          // data_size=2 (4 bytes), no multicast/pad/iter
    g1[1] = (int)(128u << 16);   // tensor_dim0 = 128
    g1[2] = (int)(98u << 16);    // tensor_dim1 = 98  (rows 98..127 read as zero)
    g1[3] = (int)(128u << 16);   // tile_dim0 = 128
    g1[4] = 128;                 // tile_dim1 = 128, tile_dim2 = 0
    g1[5] = 128;                 // tensor_dim0_stride = 128
    g1[6] = 0;
    g1[7] = 0;
    v4i gz4 = {0, 0, 0, 0};
    v8i gz8 = {0, 0, 0, 0, 0, 0, 0, 0};
    __builtin_amdgcn_tensor_load_to_lds(g0, g1, gz4, gz4, gz8, 0);
    __builtin_amdgcn_s_wait_tensorcnt(0);
  }
  __syncthreads();
  {
    const float* xf = (const float*)(smem + 32768);
    for (int i = tid; i < 128 * 128; i += 256) xs[i] = (__bf16)xf[i];
  }
  __syncthreads();

  // ---- Phase 1: qkv = xs @ W + b ; q,k row-major into qk, v transposed into vT
  for (int nt = wave * 3; nt < wave * 3 + 3; ++nt) {
    const int col = nt * 16;
    v16bf bfr[4];
#pragma unroll
    for (int kt = 0; kt < 4; ++kt) bfr[kt] = load_bT(wqT, col, 128, kt * 32);
    const float bias = qkv_b[col + nlo];
    for (int mt = 0; mt < 8; ++mt) {
      v8f acc = {};
#pragma unroll
      for (int kt = 0; kt < 4; ++kt)
        acc = wmma_bf16(load_a(xs, mt * 16, 128, kt * 32), bfr[kt], acc);
#pragma unroll
      for (int r = 0; r < 8; ++r) acc[r] += bias;
      if (col < 256) store_d_bf16(qk, mt * 16, col, 256, acc);
      else           store_d_bf16_T(vT, mt * 16, col - 256, 128, acc);
    }
  }
  __syncthreads();

  // ---- Phase 2: attention; wave w owns row tile w (7 tiles cover 112 rows) ----
  if (wave < 7) {
    const int mt = wave;
    if (lane < 16)                           // zero K-pad cols 112..127 of my p rows
#pragma unroll
      for (int r = 0; r < 16; ++r)
        pbuf[(mt * 16 + r) * 128 + 112 + lane] = (__bf16)0.0f;

    const float scale = 0.17677669529663687f;   // 1/sqrt(32)
    for (int h = 0; h < 4; ++h) {
      const __bf16* qh = qk + h * 32;
      const __bf16* kh = qk + 128 + h * 32;

      v16bf aq = load_a(qh, mt * 16, 256, 0);
      v8f s[7];
#pragma unroll
      for (int nt = 0; nt < 7; ++nt) {
        v8f z = {};
        s[nt] = wmma_bf16(aq, load_bT(kh, nt * 16, 256, 0), z);
      }
      float rmax[8];
#pragma unroll
      for (int r = 0; r < 8; ++r) rmax[r] = -1e30f;
#pragma unroll
      for (int nt = 0; nt < 7; ++nt) {
        const bool msk = (nt * 16 + nlo) >= 98;
#pragma unroll
        for (int r = 0; r < 8; ++r) {
          float v = s[nt][r] * scale;
          v = msk ? -1e30f : v;
          s[nt][r] = v;
          rmax[r] = fmaxf(rmax[r], v);
        }
      }
#pragma unroll
      for (int r = 0; r < 8; ++r)
#pragma unroll
        for (int d = 1; d < 16; d <<= 1)
          rmax[r] = fmaxf(rmax[r], __shfl_xor(rmax[r], d, 32));
      float rsum[8];
#pragma unroll
      for (int r = 0; r < 8; ++r) rsum[r] = 0.0f;
#pragma unroll
      for (int nt = 0; nt < 7; ++nt)
#pragma unroll
        for (int r = 0; r < 8; ++r) {
          float e = __expf(s[nt][r] - rmax[r]);
          s[nt][r] = e;
          rsum[r] += e;
        }
#pragma unroll
      for (int r = 0; r < 8; ++r)
#pragma unroll
        for (int d = 1; d < 16; d <<= 1)
          rsum[r] += __shfl_xor(rsum[r], d, 32);
      float rinv[8];
#pragma unroll
      for (int r = 0; r < 8; ++r) rinv[r] = 1.0f / rsum[r];
#pragma unroll
      for (int nt = 0; nt < 7; ++nt) {
        const int c = nt * 16 + nlo;
#pragma unroll
        for (int r = 0; r < 8; ++r)
          pbuf[(mt * 16 + mo + r) * 128 + c] = (__bf16)(s[nt][r] * rinv[r]);
      }
      // o = p[16x128] @ v[128x32]; v fragments contiguous from vT
      v8f o0 = {}, o1 = {};
#pragma unroll
      for (int kt = 0; kt < 4; ++kt) {
        v16bf ap = load_a(pbuf, mt * 16, 128, kt * 32);
        o0 = wmma_bf16(ap, load_bT(vT, h * 32,      128, kt * 32), o0);
        o1 = wmma_bf16(ap, load_bT(vT, h * 32 + 16, 128, kt * 32), o1);
      }
      store_d_bf16(obuf, mt * 16, h * 32,      128, o0);
      store_d_bf16(obuf, mt * 16, h * 32 + 16, 128, o1);
    }
  }
  __syncthreads();

  // ---- Phase 3: out = obuf @ proj + b ; wave w -> column tile w, 7 row tiles
  {
    const int col = wave * 16;
    v16bf bfr[4];
#pragma unroll
    for (int kt = 0; kt < 4; ++kt) bfr[kt] = load_bT(wpT, col, 128, kt * 32);
    const float bias = proj_b[col + nlo];
    for (int mt = 0; mt < 7; ++mt) {
      v8f acc = {};
#pragma unroll
      for (int kt = 0; kt < 4; ++kt)
        acc = wmma_bf16(load_a(obuf, mt * 16, 128, kt * 32), bfr[kt], acc);
#pragma unroll
      for (int r = 0; r < 8; ++r) {
        const int row = mt * 16 + mo + r;
        if (row < 98)
          out[((size_t)b * 98 + row) * 128 + col + nlo] = acc[r] + bias;
      }
    }
  }
}

extern "C" void kernel_launch(void* const* d_in, const int* in_sizes, int n_in,
                              void* d_out, int out_size, void* d_ws, size_t ws_size,
                              hipStream_t stream) {
  const float* x      = (const float*)d_in[0];
  const float* qkv_w  = (const float*)d_in[1];
  const float* qkv_b  = (const float*)d_in[2];
  const float* proj_w = (const float*)d_in[3];
  const float* proj_b = (const float*)d_in[4];
  float* out = (float*)d_out;

  __bf16* wqT = (__bf16*)d_ws;                             // [384][128] bf16
  __bf16* wpT = (__bf16*)((char*)d_ws + 128 * 384 * 2);    // [128][128] bf16

  wa3d_prep<<<192, 256, 0, stream>>>(qkv_w, proj_w, wqT, wpT);

  (void)hipFuncSetAttribute((const void*)wa3d_kernel,
                            hipFuncAttributeMaxDynamicSharedMemorySize, LDS_BYTES);
  wa3d_kernel<<<2048, 256, LDS_BYTES, stream>>>(x, qkv_b, proj_b, wqT, wpT, out);
}
